// Src_DGCNN_0_48833778155882
// MI455X (gfx1250) — matrix operand order repaired and checked by reading, not compiled, not run on hardware
//
#include <hip/hip_runtime.h>

#define KNBR 20
#define NPTS 2048
#define NB   8
#define EPS  1e-5f

typedef __bf16 bf16;
typedef __attribute__((ext_vector_type(16))) __bf16 v16bf;
typedef __attribute__((ext_vector_type(8)))  float  v8f;

// ------------------------------------------------------------------
// KNN: one wave32 per query row; per-wave distance row lives in LDS.
// ------------------------------------------------------------------
__global__ __launch_bounds__(128) void knn_kernel(const float* __restrict__ x,
                                                  int* __restrict__ idx) {
    __shared__ float dist[4][NPTS];
    const int wid  = threadIdx.x >> 5;
    const int lane = threadIdx.x & 31;
    const int row  = blockIdx.x * 4 + wid;       // row = b*NPTS + i
    const int b    = row / NPTS;
    const int i    = row - b * NPTS;

    const float xi0 = x[(b*3 + 0)*NPTS + i];
    const float xi1 = x[(b*3 + 1)*NPTS + i];
    const float xi2 = x[(b*3 + 2)*NPTS + i];

    for (int j = lane; j < NPTS; j += 32) {
        float d0 = xi0 - x[(b*3 + 0)*NPTS + j];
        float d1 = xi1 - x[(b*3 + 1)*NPTS + j];
        float d2 = xi2 - x[(b*3 + 2)*NPTS + j];
        dist[wid][j] = -(d0*d0 + d1*d1 + d2*d2);
    }
    for (int t = 0; t < KNBR; ++t) {
        float bv = -3.0e38f;
        int   bj = 0x7fffffff;
        for (int j = lane; j < NPTS; j += 32) {
            float v = dist[wid][j];
            if (v > bv) { bv = v; bj = j; }
        }
        for (int msk = 16; msk > 0; msk >>= 1) {
            float ov = __shfl_xor(bv, msk, 32);
            int   oj = __shfl_xor(bj, msk, 32);
            if (ov > bv || (ov == bv && oj < bj)) { bv = ov; bj = oj; }
        }
        if (lane == 0) {
            idx[row * KNBR + t] = bj;
            dist[wid][bj] = -3.0e38f;   // intra-wave LDS ops are in-order
        }
    }
}

// ------------------------------------------------------------------
// Edge features: ch [0..2] = x[j]-x[n], [3..5] = x[n]; bf16 [32][P]
// (rows 6..31 zero-filled by zfill_kernel so GEMM needs no K guard).
// ------------------------------------------------------------------
__global__ __launch_bounds__(256) void graphfeat_kernel(const float* __restrict__ x,
                                                        const int* __restrict__ idx,
                                                        bf16* __restrict__ h0, int P) {
    int p = blockIdx.x * blockDim.x + threadIdx.x;
    if (p >= P) return;
    int t = p / KNBR;
    int n = t % NPTS;
    int b = t / NPTS;
    int j = idx[p];
#pragma unroll
    for (int c = 0; c < 3; ++c) {
        float xc = x[(b*3 + c)*NPTS + n];
        float xj = x[(b*3 + c)*NPTS + j];
        h0[(size_t)c       * P + p] = (bf16)(xj - xc);
        h0[(size_t)(3 + c) * P + p] = (bf16)xc;
    }
}

__global__ __launch_bounds__(256) void zfill_kernel(unsigned int* __restrict__ p, int n32) {
    int i = blockIdx.x * blockDim.x + threadIdx.x;
    if (i < n32) p[i] = 0u;
}

__global__ void zero_stats(float* __restrict__ stats) {
    stats[threadIdx.x] = 0.0f;   // 1 block, 1024 threads
}

// ------------------------------------------------------------------
// Pack fp32 weights (O x I) into bf16 WMMA-A-fragment order, K padded
// to Ipad (multiple of 32). Layout: ((otile*nk + kb)*32 + lane)*16 + e
// lane<16 : M = otile*16+lane,    K = kb*32 + e (e<8) / e+8 (e>=8)
// lane>=16: same M,               K += 8
// ------------------------------------------------------------------
__global__ __launch_bounds__(256) void pack_w_kernel(const float* __restrict__ W,
                                                     bf16* __restrict__ Wb,
                                                     int O, int I, int Ipad) {
    int gid = blockIdx.x * blockDim.x + threadIdx.x;
    int total = O * Ipad;
    if (gid >= total) return;
    int nk    = Ipad >> 5;
    int e     = gid & 15;
    int lane  = (gid >> 4) & 31;
    int kb    = (gid >> 9) % nk;
    int otile = (gid >> 9) / nk;
    int m     = otile * 16 + (lane & 15);
    int half  = lane >> 4;
    int kk    = kb * 32 + half * 8 + ((e < 8) ? e : (e + 8));
    Wb[gid] = (kk < I) ? (bf16)W[m * I + kk] : (bf16)0.0f;
}

// ------------------------------------------------------------------
// GEMM y[O x P] = Wb * act[Ipad x P], fused BN-stat reduction.
// One 16x16 tile per wave; branch-free K loop: one 32B A load,
// 16 strided 2B B loads, one v_wmma per step.
// ------------------------------------------------------------------
__global__ __launch_bounds__(256) void gemm_bn_kernel(const bf16* __restrict__ Wb,
                                                      const bf16* __restrict__ act,
                                                      bf16*  __restrict__ y,
                                                      float* __restrict__ stats, // [0..511]=sum [512..1023]=sumsq
                                                      int Ipad, int P) {
    const int lane  = threadIdx.x & 31;
    const int wave  = threadIdx.x >> 5;
    const int half  = lane >> 4;
    const int r     = lane & 15;
    const int pbase = (blockIdx.x * 8 + wave) * 16;        // position tile
    const int nk    = Ipad >> 5;

    const bf16* wp = Wb + ((size_t)(blockIdx.y * nk) * 32 + lane) * 16;
    const bf16* ap = act + (size_t)(8 * half) * P + pbase + r;

    v8f acc = {0.f,0.f,0.f,0.f,0.f,0.f,0.f,0.f};
    for (int kb = 0; kb < nk; ++kb) {
        v16bf a   = *(const v16bf*)(wp);
        v16bf bfr;
#pragma unroll
        for (int s = 0; s < 16; ++s) bfr[s] = ap[(size_t)((s < 8) ? s : (s + 8)) * P];
        acc = __builtin_amdgcn_wmma_f32_16x16x32_bf16(
                  false, a, false, bfr, (short)0, acc, false, false);
        wp += 512;                      // next 32-lane fragment block
        ap += (size_t)32 * P;           // next 32 K rows
    }

    // Epilogue: store bf16 pre-norm y, reduce per-channel sum/sumsq.
#pragma unroll
    for (int v = 0; v < 8; ++v) {
        const int row = blockIdx.y * 16 + v + 8 * half;    // D: M = v + 8*half, N = lane&15
        float val = acc[v];
        y[(size_t)row * P + pbase + r] = (bf16)val;
        float s = val, q = val * val;
#pragma unroll
        for (int mm = 1; mm < 16; mm <<= 1) {              // reduce across the 16-lane half
            s += __shfl_xor(s, mm, 32);
            q += __shfl_xor(q, mm, 32);
        }
        if (r == 0) {
            atomicAdd(&stats[row], s);
            atomicAdd(&stats[512 + row], q);
        }
    }
}

__global__ void bn_finalize(const float* __restrict__ stats,
                            const float* __restrict__ g, const float* __restrict__ bt,
                            float* __restrict__ sc, float* __restrict__ sh,
                            int O, float invP) {
    int c = blockIdx.x * blockDim.x + threadIdx.x;
    if (c >= O) return;
    float mean = stats[c] * invP;
    float var  = stats[512 + c] * invP - mean * mean;      // ddof=0, matches jnp.var
    float s    = g[c] * rsqrtf(var + EPS);
    sc[c] = s;
    sh[c] = bt[c] - mean * s;
}

// ------------------------------------------------------------------
// Affine + ReLU + (optional) next-layer act + max over k into concat.
// ------------------------------------------------------------------
__global__ __launch_bounds__(256) void ewise_kernel(const bf16* __restrict__ y,
                                                    const float* __restrict__ sc,
                                                    const float* __restrict__ sh,
                                                    bf16* __restrict__ act_out,   // may be null
                                                    bf16* __restrict__ xcat,      // [512][B*N]
                                                    int O, int P, int R, int choff) {
    int gid = blockIdx.x * blockDim.x + threadIdx.x;
    if (gid >= O * R) return;
    int c = gid / R;
    int r = gid - c * R;
    float s = sc[c], t = sh[c];
    size_t base = (size_t)c * P + (size_t)r * KNBR;
    float mv = -3.0e38f;
    if (act_out) {
#pragma unroll
        for (int kk = 0; kk < KNBR; ++kk) {
            float f = fmaxf((float)y[base + kk] * s + t, 0.0f);
            act_out[base + kk] = (bf16)f;
            mv = fmaxf(mv, f);
        }
    } else {
#pragma unroll
        for (int kk = 0; kk < KNBR; ++kk) {
            float f = fmaxf((float)y[base + kk] * s + t, 0.0f);
            mv = fmaxf(mv, f);
        }
    }
    xcat[(size_t)(choff + c) * R + r] = (bf16)mv;
}

// conv5 epilogue: affine + ReLU -> fp32 output (B, 512, N)
__global__ __launch_bounds__(256) void ewise_final(const bf16* __restrict__ y,
                                                   const float* __restrict__ sc,
                                                   const float* __restrict__ sh,
                                                   float* __restrict__ out, int R) {
    int gid = blockIdx.x * blockDim.x + threadIdx.x;
    if (gid >= 512 * R) return;
    int c = gid / R;
    int r = gid - c * R;
    float f = fmaxf((float)y[(size_t)c * R + r] * sc[c] + sh[c], 0.0f);
    int b = r / NPTS;
    int n = r - b * NPTS;
    out[((size_t)b * 512 + c) * NPTS + n] = f;
}

extern "C" void kernel_launch(void* const* d_in, const int* in_sizes, int n_in,
                              void* d_out, int out_size, void* d_ws, size_t ws_size,
                              hipStream_t stream) {
    const float* x  = (const float*)d_in[0];
    const float* w1 = (const float*)d_in[1];  const float* g1 = (const float*)d_in[2];  const float* b1 = (const float*)d_in[3];
    const float* w2 = (const float*)d_in[4];  const float* g2 = (const float*)d_in[5];  const float* b2 = (const float*)d_in[6];
    const float* w3 = (const float*)d_in[7];  const float* g3 = (const float*)d_in[8];  const float* b3 = (const float*)d_in[9];
    const float* w4 = (const float*)d_in[10]; const float* g4 = (const float*)d_in[11]; const float* b4 = (const float*)d_in[12];
    const float* w5 = (const float*)d_in[13]; const float* g5 = (const float*)d_in[14]; const float* b5 = (const float*)d_in[15];

    const int P = NB * NPTS * KNBR;   // 327,680 edge positions
    const int R = NB * NPTS;          // 16,384 point positions

    // ---- carve workspace (256B aligned) ----
    char* ws = (char*)d_ws;
    size_t off = 0;
    auto carve = [&](size_t bytes) { void* p = ws + off; off += (bytes + 255) & ~(size_t)255; return p; };
    int*   idx   = (int*)  carve((size_t)P * sizeof(int));
    bf16*  act0  = (bf16*) carve((size_t)32  * P * sizeof(bf16));   // rows 6..31 zero
    bf16*  act1  = (bf16*) carve((size_t)64  * P * sizeof(bf16));
    bf16*  act2  = (bf16*) carve((size_t)64  * P * sizeof(bf16));
    bf16*  act3  = (bf16*) carve((size_t)128 * P * sizeof(bf16));
    bf16*  xcat  = (bf16*) carve((size_t)512 * R * sizeof(bf16));
    bf16*  ybuf  = (bf16*) carve((size_t)256 * P * sizeof(bf16));   // largest y; reused by all layers
    bf16*  wb1   = (bf16*) carve((size_t) 64 *  32 * sizeof(bf16));
    bf16*  wb2   = (bf16*) carve((size_t) 64 *  64 * sizeof(bf16));
    bf16*  wb3   = (bf16*) carve((size_t)128 *  64 * sizeof(bf16));
    bf16*  wb4   = (bf16*) carve((size_t)256 * 128 * sizeof(bf16));
    bf16*  wb5   = (bf16*) carve((size_t)512 * 512 * sizeof(bf16));
    float* stats = (float*)carve(1024 * sizeof(float));
    float* scale = (float*)carve(512 * sizeof(float));
    float* shift = (float*)carve(512 * sizeof(float));

    // ---- weight packing (bf16, WMMA A-fragment order, K padded) ----
    auto pack = [&](const float* W, bf16* Wb, int O, int I, int Ipad) {
        pack_w_kernel<<<((O * Ipad) + 255) / 256, 256, 0, stream>>>(W, Wb, O, I, Ipad);
    };
    pack(w1, wb1,  64,   6,  32);
    pack(w2, wb2,  64,  64,  64);
    pack(w3, wb3, 128,  64,  64);
    pack(w4, wb4, 256, 128, 128);
    pack(w5, wb5, 512, 512, 512);

    // ---- KNN + graph feature (+ zero pad rows of act0) ----
    knn_kernel<<<(NB * NPTS) / 4, 128, 0, stream>>>(x, idx);
    {
        int n32 = (26 * P) / 2;   // rows 6..31 of act0, as dwords
        zfill_kernel<<<(n32 + 255) / 256, 256, 0, stream>>>((unsigned int*)(act0 + (size_t)6 * P), n32);
    }
    graphfeat_kernel<<<(P + 255) / 256, 256, 0, stream>>>(x, idx, act0, P);

    // ---- generic conv+BN+ReLU(+max) layer driver ----
    auto layer = [&](int O, int Ipad, int Pl, const bf16* Wb, const float* g, const float* bt,
                     const bf16* in, bf16* actOut, int choff) {
        zero_stats<<<1, 1024, 0, stream>>>(stats);
        dim3 grid(Pl / 128, O / 16);
        gemm_bn_kernel<<<grid, 256, 0, stream>>>(Wb, in, ybuf, stats, Ipad, Pl);
        bn_finalize<<<(O + 255) / 256, 256, 0, stream>>>(stats, g, bt, scale, shift, O, 1.0f / (float)Pl);
        ewise_kernel<<<((O * R) + 255) / 256, 256, 0, stream>>>(ybuf, scale, shift, actOut, xcat, O, Pl, R, choff);
    };

    layer( 64,  32, P, wb1, g1, b1, act0, act1,   0);
    layer( 64,  64, P, wb2, g2, b2, act1, act2,  64);
    layer(128,  64, P, wb3, g3, b3, act2, act3, 128);
    layer(256, 128, P, wb4, g4, b4, act3, nullptr, 256);   // only max kept

    // ---- conv5: 512 -> 512 on (B*N) positions ----
    zero_stats<<<1, 1024, 0, stream>>>(stats);
    dim3 grid5(R / 128, 512 / 16);
    gemm_bn_kernel<<<grid5, 256, 0, stream>>>(wb5, xcat, ybuf, stats, 512, R);
    bn_finalize<<<2, 256, 0, stream>>>(stats, g5, b5, scale, shift, 512, 1.0f / (float)R);
    ewise_final<<<((512 * R) + 255) / 256, 256, 0, stream>>>(ybuf, scale, shift, (float*)d_out, R);
}
